// NNConvNet_55155970015707
// MI455X (gfx1250) — matrix operand order repaired and checked by reading, not compiled, not run on hardware
//
#include <hip/hip_runtime.h>
#include <hip/hip_bf16.h>

// ---------------------------------------------------------------------------
// NNConv(edge-network) x2 + Set2Set + MLP head for MI455X (gfx1250, wave32).
//
// Dominant compute: per-edge h(128) @ w_e2(128x576) = 29.5 GFLOP/layer, run on
// v_wmma_f32_16x16x32_f16 (f32 accumulate). Weights are pre-swizzled into
// WMMA B-fragment order so each lane's v16h operand is a single contiguous
// 32-byte global load (coalesced, WGP$/L2 resident: swizzled w_e2 = 147 KB).
// The h tile is staged in LDS already in A-fragment order -> ds_load_b128.
// Scatter-mean uses global_atomic_add_f32 into an L2-resident 4.8 MB buffer.
// ---------------------------------------------------------------------------

typedef __attribute__((ext_vector_type(16))) _Float16 v16h;
typedef __attribute__((ext_vector_type(8)))  float    v8f;
typedef __attribute__((ext_vector_type(4)))  float    v4f;

#define N_NODES 50000
#define N_EDGES 200000
#define N_GRAPHS 1024
#define DD 24          // node feature dim
#define E_FEAT 8
#define E_HID 128
#define DSQ (DD * DD)  // 576

// LDS per wave: h tile in A-fragment order + 16x576 W tile (padded rows)
#define H_FRAG_BYTES 4096               // 4 kt x 32 lanes x 16 halves x 2B
#define WSTRIDE 580                     // floats per W row (576 + pad)
#define W_BYTES (16 * WSTRIDE * 4)      // 37120 B
#define WAVE_BYTES (H_FRAG_BYTES + W_BYTES)
#define WAVES_PER_WG 2

// ---- fragment index maps (CDNA5 ISA §7.12.2, wave32) ------------
// A (16-bit, 16x32): lane group g=lane>>4, row m=lane&15;
//   half slot hi -> k = hi + (hi<8 ? 0 : 8) + 8*g
// B (16-bit, 32x16): col n=lane&15; w_e2f uses the same k map as A;
//   w_e1f uses hi -> k = hi + 16*g (equals A's map for the real k<8)
// C/D (f32 16x16): vgpr r -> D[m = r + 8*g][n = lane&15]

// ---------------------------------------------------------------------------
__global__ void fill_f32(float* p, float v, int n) {
    int i = blockIdx.x * blockDim.x + threadIdx.x;
    if (i < n) p[i] = v;
}
__global__ void fill_u32(unsigned* p, unsigned v, int n) {
    int i = blockIdx.x * blockDim.x + threadIdx.x;
    if (i < n) p[i] = v;
}

// One-time weight swizzle into fragment-major order:
//  w_e1f[nt][lane][hi]      (8 N-tiles,  K zero-padded 8->32)   = 8 KB
//  w_e2f[nt*4+kt][lane][hi] (36 N-tiles, 4 K-tiles of 32)       = 147 KB
__global__ void prep_weights(const float* __restrict__ w_e1,
                             const float* __restrict__ w_e2,
                             _Float16* __restrict__ w_e1f,
                             _Float16* __restrict__ w_e2f) {
    int i = blockIdx.x * blockDim.x + threadIdx.x;
    if (i < 8 * 32 * 16) {
        int nt = i >> 9, rem = i & 511, lane = rem >> 4, hi = rem & 15;
        int g = lane >> 4, n = lane & 15;
        int k = hi + 16 * g;                    // B-frag K within 0..31
        int col = nt * 16 + n;
        w_e1f[i] = (_Float16)(k < E_FEAT ? w_e1[k * E_HID + col] : 0.0f);
    } else {
        int j = i - 8 * 32 * 16;
        if (j < 36 * 4 * 32 * 16) {
            int t = j >> 9, rem = j & 511, lane = rem >> 4, hi = rem & 15;
            int nt = t >> 2, kt = t & 3;
            int g = lane >> 4, n = lane & 15;
            int k = kt * 32 + (hi & 7) + ((hi & 8) ? 16 : 0) + 8 * g;
            int col = nt * 16 + n;
            w_e2f[j] = (_Float16)w_e2[k * DSQ + col];
        }
    }
}

// ---------------------------------------------------------------------------
// Fused edge kernel: per wave = one 16-edge tile.
//   1) h = relu(edge_attr @ w_e1 + b_e1)        8  x v_wmma_f32_16x16x32_f16
//   2) W = h @ w_e2 + b_e2  (16x576 -> LDS)   144  x v_wmma_f32_16x16x32_f16
//   3) msg[e,o] = sum_i x[src_e,i] * W[e,i*24+o]   (fp32 VALU, b128 LDS reads)
//   4) atomicAdd into per-node sum + count
// ---------------------------------------------------------------------------
__global__ __launch_bounds__(WAVES_PER_WG * 32)
void edge_msg_kernel(const float* __restrict__ x,            // [N_NODES, 24]
                     const float* __restrict__ edge_attr,    // [E, 8]
                     const int*   __restrict__ edge_index,   // [2, E]
                     const _Float16* __restrict__ w_e1f,     // frag order
                     const float* __restrict__ b_e1,         // [128]
                     const _Float16* __restrict__ w_e2f,     // frag order
                     const float* __restrict__ b_e2,         // [576]
                     float* __restrict__ sum_out,            // [N_NODES, 24]
                     float* __restrict__ cnt_out) {          // [N_NODES]
    extern __shared__ char smem[];
    const int wave = threadIdx.x >> 5;
    const int lane = threadIdx.x & 31;
    const int g    = lane >> 4;
    const int mrow = lane & 15;
    const int tile = blockIdx.x * WAVES_PER_WG + wave;
    const int e0   = tile * 16;
    if (e0 >= N_EDGES) return;

    char* base = smem + wave * WAVE_BYTES;
    _Float16* h_frag = (_Float16*)base;             // [4][32][16] halves
    float*    w_lds  = (float*)(base + H_FRAG_BYTES); // [16][WSTRIDE]

    __builtin_prefetch(edge_attr + (size_t)(e0 + 16) * E_FEAT, 0, 0);

    // ---- A fragment: edge_attr tile, K zero-padded 8 -> 32 -----------------
    // Per the A map, real K<8 lives only in lanes 0-15 half-slots 0..7.
    v16h aEA = {};
    if (g == 0) {
        const v4f* row = (const v4f*)(edge_attr + (size_t)(e0 + mrow) * E_FEAT);
        v4f lo = row[0], hi4 = row[1];
#pragma unroll
        for (int j = 0; j < 4; ++j) {
            aEA[j]     = (_Float16)lo[j];
            aEA[4 + j] = (_Float16)hi4[j];
        }
    }

    // ---- h = relu(ea @ w_e1 + b_e1), written to LDS in A-fragment order ----
#pragma unroll
    for (int nt = 0; nt < E_HID / 16; ++nt) {
        v16h bF = *(const v16h*)(w_e1f + ((nt * 32 + lane) << 4));
        v8f c = {};
        c = __builtin_amdgcn_wmma_f32_16x16x32_f16(false, aEA, false, bF,
                                                   (short)0, c, false, false);
#pragma unroll
        for (int r = 0; r < 8; ++r) {
            int m = r + 8 * g;
            int k = nt * 16 + mrow;             // hidden index = K of big GEMM
            float v = fmaxf(c[r] + b_e1[k], 0.0f);
            int kt2 = k >> 5, kk = k & 31;      // inverse A-frag map
            int g2  = (kk >> 3) & 1;
            int hi2 = (kk & 7) + ((kk & 16) ? 8 : 0);
            h_frag[((kt2 * 32 + g2 * 16 + m) << 4) + hi2] = (_Float16)v;
        }
    }

    // ---- A fragments: one 32B contiguous LDS read each ---------------------
    v16h hA[4];
#pragma unroll
    for (int kt = 0; kt < 4; ++kt)
        hA[kt] = *(const v16h*)(h_frag + ((kt * 32 + lane) << 4));

    // ---- W = h @ w_e2 + b_e2 : 36 N-tiles x 4 K-steps ----------------------
    for (int nt = 0; nt < DSQ / 16; ++nt) {
        v8f acc = {};
#pragma unroll
        for (int kt = 0; kt < 4; ++kt) {
            v16h bF = *(const v16h*)(w_e2f + (((nt * 4 + kt) * 32 + lane) << 4));
            acc = __builtin_amdgcn_wmma_f32_16x16x32_f16(
                false, hA[kt], false, bF, (short)0, acc, false, false);
        }
        int col = nt * 16 + mrow;
        float bias = b_e2[col];
#pragma unroll
        for (int r = 0; r < 8; ++r)
            w_lds[(r + 8 * g) * WSTRIDE + col] = acc[r] + bias;
    }

    // ---- per-edge matvec + scatter-add (lane: edge mrow, outs g*12..+11) ---
    const int e    = e0 + mrow;
    const int srcn = edge_index[e];
    const int dstn = edge_index[N_EDGES + e];
    const v4f* xrow = (const v4f*)(x + (size_t)srcn * DD);   // 96B rows, 16B aligned
    float xs[DD];
#pragma unroll
    for (int q = 0; q < 6; ++q) {
        v4f t = xrow[q];
#pragma unroll
        for (int j = 0; j < 4; ++j) xs[4 * q + j] = t[j];
    }
    float msg[12] = {0, 0, 0, 0, 0, 0, 0, 0, 0, 0, 0, 0};
#pragma unroll
    for (int i = 0; i < DD; ++i) {
        const v4f* wrow = (const v4f*)(w_lds + mrow * WSTRIDE + i * DD + g * 12);
#pragma unroll
        for (int q = 0; q < 3; ++q) {
            v4f w4 = wrow[q];
#pragma unroll
            for (int j = 0; j < 4; ++j) msg[4 * q + j] += xs[i] * w4[j];
        }
    }
#pragma unroll
    for (int oo = 0; oo < 12; ++oo)
        atomicAdd(&sum_out[(size_t)dstn * DD + g * 12 + oo], msg[oo]);
    if (g == 0) atomicAdd(&cnt_out[dstn], 1.0f);
}

// ---------------------------------------------------------------------------
// node combine: x_out = sum/max(cnt,1) + x_in @ root + bias  (+ optional relu)
// ---------------------------------------------------------------------------
__global__ void combine_kernel(const float* __restrict__ sum_in,
                               const float* __restrict__ cnt,
                               const float* __restrict__ x_in,
                               const float* __restrict__ root,   // [24,24]
                               const float* __restrict__ bias,   // [24]
                               float* __restrict__ x_out, int do_relu) {
    int idx = blockIdx.x * blockDim.x + threadIdx.x;
    if (idx >= N_NODES * DD) return;
    int n = idx / DD, o = idx % DD;
    float agg = sum_in[idx] / fmaxf(cnt[n], 1.0f);
    float d = 0.0f;
#pragma unroll
    for (int i = 0; i < DD; ++i) d += x_in[n * DD + i] * root[i * DD + o];
    float v = agg + d + bias[o];
    if (do_relu) v = fmaxf(v, 0.0f);
    x_out[idx] = v;
}

// ---------------------------------------------------------------------------
// Set2Set pieces
// ---------------------------------------------------------------------------
__device__ __forceinline__ unsigned enc_f(float f) {
    unsigned u = __float_as_uint(f);
    return (u & 0x80000000u) ? ~u : (u | 0x80000000u);   // monotone key
}
__device__ __forceinline__ float dec_f(unsigned k) {
    unsigned u = (k & 0x80000000u) ? (k & 0x7fffffffu) : ~k;
    return __uint_as_float(u);
}
__device__ __forceinline__ float sigf(float x) { return 1.0f / (1.0f + __expf(-x)); }

__global__ __launch_bounds__(96)
void lstm_kernel(const float* __restrict__ q_star,   // [B,48]
                 const float* __restrict__ w_ih,     // [96,48]
                 const float* __restrict__ w_hh,     // [96,24]
                 const float* __restrict__ b_ih, const float* __restrict__ b_hh,
                 float* __restrict__ hbuf, float* __restrict__ cbuf) {
    __shared__ float gates[96];
    int b = blockIdx.x, j = threadIdx.x;
    float s = b_ih[j] + b_hh[j];
#pragma unroll
    for (int k = 0; k < 2 * DD; ++k) s += w_ih[j * 2 * DD + k] * q_star[b * 2 * DD + k];
#pragma unroll
    for (int k = 0; k < DD; ++k) s += w_hh[j * DD + k] * hbuf[b * DD + k];
    gates[j] = s;
    __syncthreads();
    if (j < DD) {
        float gi = sigf(gates[j]);
        float gf = sigf(gates[DD + j]);
        float gg = tanhf(gates[2 * DD + j]);
        float go = sigf(gates[3 * DD + j]);
        float cn = gf * cbuf[b * DD + j] + gi * gg;
        cbuf[b * DD + j] = cn;
        hbuf[b * DD + j] = go * tanhf(cn);   // q = h
    }
}

__global__ void attn_score_kernel(const float* __restrict__ x,     // [N,24]
                                  const float* __restrict__ q,     // [B,24]
                                  const int* __restrict__ batch,
                                  float* __restrict__ e_node,
                                  unsigned* __restrict__ emax) {
    int n = blockIdx.x * blockDim.x + threadIdx.x;
    if (n >= N_NODES) return;
    int b = batch[n];
    float e = 0.0f;
#pragma unroll
    for (int i = 0; i < DD; ++i) e += x[n * DD + i] * q[b * DD + i];
    e_node[n] = e;
    atomicMax(&emax[b], enc_f(e));
}

__global__ void attn_acc_kernel(const float* __restrict__ x,
                                const int* __restrict__ batch,
                                const float* __restrict__ e_node,
                                const unsigned* __restrict__ emax,
                                float* __restrict__ denom,
                                float* __restrict__ rbuf) {
    int idx = blockIdx.x * blockDim.x + threadIdx.x;
    if (idx >= N_NODES * DD) return;
    int n = idx / DD, o = idx % DD, b = batch[n];
    float ex = __expf(e_node[n] - dec_f(emax[b]));
    if (o == 0) atomicAdd(&denom[b], ex);
    atomicAdd(&rbuf[b * DD + o], ex * x[idx]);
}

__global__ void qstar_kernel(const float* __restrict__ hbuf,
                             const float* __restrict__ rbuf,
                             const float* __restrict__ denom,
                             float* __restrict__ q_star) {
    int idx = blockIdx.x * blockDim.x + threadIdx.x;
    if (idx >= N_GRAPHS * 2 * DD) return;
    int b = idx / (2 * DD), j = idx % (2 * DD);
    q_star[idx] = (j < DD) ? hbuf[b * DD + j]
                           : rbuf[b * DD + (j - DD)] / fmaxf(denom[b], 1e-16f);
}

__global__ void head_kernel(const float* __restrict__ q_star,
                            const float* __restrict__ w_fc2, const float* __restrict__ b_fc2,
                            const float* __restrict__ w_fc3, const float* __restrict__ b_fc3,
                            float* __restrict__ out) {
    int b = blockIdx.x * blockDim.x + threadIdx.x;
    if (b >= N_GRAPHS) return;
    float hid[8];
#pragma unroll
    for (int u = 0; u < 8; ++u) {
        float s = b_fc2[u];
#pragma unroll
        for (int k = 0; k < 2 * DD; ++k) s += q_star[b * 2 * DD + k] * w_fc2[k * 8 + u];
        hid[u] = fmaxf(s, 0.0f);
    }
#pragma unroll
    for (int v = 0; v < 2; ++v) {
        float s = b_fc3[v];
#pragma unroll
        for (int u = 0; u < 8; ++u) s += hid[u] * w_fc3[u * 2 + v];
        out[b * 2 + v] = s;
    }
}

// ---------------------------------------------------------------------------
static inline size_t al256(size_t x) { return (x + 255) & ~(size_t)255; }

extern "C" void kernel_launch(void* const* d_in, const int* in_sizes, int n_in,
                              void* d_out, int out_size, void* d_ws, size_t ws_size,
                              hipStream_t stream) {
    (void)in_sizes; (void)n_in; (void)out_size; (void)ws_size;
    const float* x0        = (const float*)d_in[0];
    const float* edge_attr = (const float*)d_in[1];
    const float* w_e1      = (const float*)d_in[2];
    const float* b_e1      = (const float*)d_in[3];
    const float* w_e2      = (const float*)d_in[4];
    const float* b_e2      = (const float*)d_in[5];
    const float* root      = (const float*)d_in[6];
    const float* bias_conv = (const float*)d_in[7];
    const float* w_ih      = (const float*)d_in[8];
    const float* w_hh      = (const float*)d_in[9];
    const float* b_ih      = (const float*)d_in[10];
    const float* b_hh      = (const float*)d_in[11];
    const float* w_fc2     = (const float*)d_in[12];
    const float* b_fc2     = (const float*)d_in[13];
    const float* w_fc3     = (const float*)d_in[14];
    const float* b_fc3     = (const float*)d_in[15];
    const int*   edge_index= (const int*)d_in[16];
    const int*   batch     = (const int*)d_in[17];
    float* out = (float*)d_out;

    // ---- workspace carve-up (~15.4 MB) ------------------------------------
    char* ws = (char*)d_ws; size_t off = 0;
    _Float16* w_e1f = (_Float16*)(ws + off); off = al256(off + 8 * 32 * 16 * 2);
    _Float16* w_e2f = (_Float16*)(ws + off); off = al256(off + (size_t)36 * 4 * 32 * 16 * 2);
    float* sum_buf  = (float*)(ws + off);    off = al256(off + (size_t)N_NODES * DD * 4);
    float* cnt_buf  = (float*)(ws + off);    off = al256(off + (size_t)N_NODES * 4);
    float* x1       = (float*)(ws + off);    off = al256(off + (size_t)N_NODES * DD * 4);
    float* x2       = (float*)(ws + off);    off = al256(off + (size_t)N_NODES * DD * 4);
    float* e_node   = (float*)(ws + off);    off = al256(off + (size_t)N_NODES * 4);
    unsigned* emax  = (unsigned*)(ws + off); off = al256(off + (size_t)N_GRAPHS * 4);
    float* denom    = (float*)(ws + off);    off = al256(off + (size_t)N_GRAPHS * 4);
    float* rbuf     = (float*)(ws + off);    off = al256(off + (size_t)N_GRAPHS * DD * 4);
    float* hbuf     = (float*)(ws + off);    off = al256(off + (size_t)N_GRAPHS * DD * 4);
    float* cbuf     = (float*)(ws + off);    off = al256(off + (size_t)N_GRAPHS * DD * 4);
    float* q_star   = (float*)(ws + off);    off = al256(off + (size_t)N_GRAPHS * 2 * DD * 4);

    const int T = 256;
    auto blocks = [](int n, int t) { return (n + t - 1) / t; };

    // weight swizzle (one pass)
    prep_weights<<<blocks(8 * 32 * 16 + 36 * 4 * 32 * 16, T), T, 0, stream>>>(
        w_e1, w_e2, w_e1f, w_e2f);

    const int edge_grid  = N_EDGES / (16 * WAVES_PER_WG);
    const size_t edge_sh = (size_t)WAVES_PER_WG * WAVE_BYTES;

    // ---- NNConv layer 1 ----------------------------------------------------
    fill_f32<<<blocks(N_NODES * DD, T), T, 0, stream>>>(sum_buf, 0.0f, N_NODES * DD);
    fill_f32<<<blocks(N_NODES, T), T, 0, stream>>>(cnt_buf, 0.0f, N_NODES);
    edge_msg_kernel<<<edge_grid, WAVES_PER_WG * 32, edge_sh, stream>>>(
        x0, edge_attr, edge_index, w_e1f, b_e1, w_e2f, b_e2, sum_buf, cnt_buf);
    combine_kernel<<<blocks(N_NODES * DD, T), T, 0, stream>>>(
        sum_buf, cnt_buf, x0, root, bias_conv, x1, /*relu=*/1);

    // ---- NNConv layer 2 ----------------------------------------------------
    fill_f32<<<blocks(N_NODES * DD, T), T, 0, stream>>>(sum_buf, 0.0f, N_NODES * DD);
    fill_f32<<<blocks(N_NODES, T), T, 0, stream>>>(cnt_buf, 0.0f, N_NODES);
    edge_msg_kernel<<<edge_grid, WAVES_PER_WG * 32, edge_sh, stream>>>(
        x1, edge_attr, edge_index, w_e1f, b_e1, w_e2f, b_e2, sum_buf, cnt_buf);
    combine_kernel<<<blocks(N_NODES * DD, T), T, 0, stream>>>(
        sum_buf, cnt_buf, x1, root, bias_conv, x2, /*relu=*/0);

    // ---- Set2Set (3 processing steps) --------------------------------------
    fill_f32<<<blocks(N_GRAPHS * DD, T), T, 0, stream>>>(hbuf, 0.0f, N_GRAPHS * DD);
    fill_f32<<<blocks(N_GRAPHS * DD, T), T, 0, stream>>>(cbuf, 0.0f, N_GRAPHS * DD);
    fill_f32<<<blocks(N_GRAPHS * 2 * DD, T), T, 0, stream>>>(q_star, 0.0f, N_GRAPHS * 2 * DD);

    for (int step = 0; step < 3; ++step) {
        lstm_kernel<<<N_GRAPHS, 96, 0, stream>>>(q_star, w_ih, w_hh, b_ih, b_hh, hbuf, cbuf);
        fill_u32<<<blocks(N_GRAPHS, T), T, 0, stream>>>(emax, 0u, N_GRAPHS);
        fill_f32<<<blocks(N_GRAPHS, T), T, 0, stream>>>(denom, 0.0f, N_GRAPHS);
        fill_f32<<<blocks(N_GRAPHS * DD, T), T, 0, stream>>>(rbuf, 0.0f, N_GRAPHS * DD);
        attn_score_kernel<<<blocks(N_NODES, T), T, 0, stream>>>(x2, hbuf, batch, e_node, emax);
        attn_acc_kernel<<<blocks(N_NODES * DD, T), T, 0, stream>>>(x2, batch, e_node, emax, denom, rbuf);
        qstar_kernel<<<blocks(N_GRAPHS * 2 * DD, T), T, 0, stream>>>(hbuf, rbuf, denom, q_star);
    }

    // ---- head ---------------------------------------------------------------
    head_kernel<<<blocks(N_GRAPHS, T), T, 0, stream>>>(q_star, w_fc2, b_fc2, w_fc3, b_fc3, out);
}
